// PwcNet_Single_25838523253068
// MI455X (gfx1250) — compile-verified, matrix operand order checked by reading, not run on hardware
//
#include <hip/hip_runtime.h>
#include <hip/hip_bf16.h>

// ---------------------------------------------------------------------------
// PWC-Net single-direction disparity pyramid for MI455X (gfx1250, wave32).
// Convs = implicit GEMM on v_wmma_f32_16x16x32_f16:
//   M = pixels, N = Cout, K = 9*Cin (Cin padded to 32 per tap chunk).
// Each wave computes a 16(pixel) x NJ*16(cout) register tile where NJ is a
// compile-time template parameter (1..4): per (tap,kchunk) it issues
// 2x global_load_b128 (A fragment) + 2*NJ x b128 (B fragments, hoisted into
// one load clause) followed by NJ back-to-back WMMAs -> loads overlap WMMA
// issue via partial loadcnt waits; accumulators stay pinned in place.
// Weights are pre-packed in exact B-fragment order; activations are packed
// f16 [P][C] with C a multiple of 32, so concat pairs never straddle a
// K-chunk (wave-uniform source select).
// ---------------------------------------------------------------------------

typedef __attribute__((ext_vector_type(16))) _Float16 v16h;
typedef __attribute__((ext_vector_type(8)))  _Float16 v8h;
typedef __attribute__((ext_vector_type(8)))  float    v8f;

#define BATCH 8

// K index held by element e (0..15) of a 16-bit WMMA fragment for lane half hi.
__device__ __forceinline__ int wmma_kmap(int e, int hi) {
    int vg = e >> 1, pr = e & 1;
    int k = (vg < 4) ? (vg * 2 + pr) : (16 + (vg - 4) * 2 + pr);
    return k + hi * 8;
}

// Load one 16-half WMMA fragment as two b128 loads.
__device__ __forceinline__ v16h ldfrag(const _Float16* __restrict__ p) {
    v8h lo = *(const v8h*)p;
    v8h hi = *(const v8h*)(p + 8);
    return __builtin_shufflevector(lo, hi,
            0, 1, 2, 3, 4, 5, 6, 7, 8, 9, 10, 11, 12, 13, 14, 15);
}

// Epilogue: bias + activation + store one 16x16 D tile.
__device__ __forceinline__ void conv_store(
    v8f acc, int ntj, int L, int mt, int P, int Cout,
    const float* __restrict__ bias, _Float16* __restrict__ outh,
    float* __restrict__ outf, int act)
{
    const int n = L & 15, hi = L >> 4;
    const int co = ntj * 16 + n;
    if (co >= Cout) return;
    const float bv = bias[co];
#pragma unroll
    for (int r = 0; r < 8; ++r) {
        const int p = mt * 16 + r + hi * 8;
        if (p < P) {
            float v = acc[r] + bv;
            if (act) v = (v > 0.0f) ? v : 0.1f * v;
            if (outh) outh[(size_t)p * Cout + co] = (_Float16)v;
            else      outf[p] = v;
        }
    }
}

// ---------------------------------------------------------------------------
// Implicit-GEMM 3x3 conv, SAME padding, optional dilation.
// act0/act1: packed f16 activations (Ca0, Ca1 channels, each multiple of 32).
// wfrag: f16 weights in fragment order [tap][kc][nt][lane(32)][e(16)].
// Output: f16 packed (outh, stride Cout) or f32 (outf, Cout==1 flow head).
// act: 0 = linear, 1 = leaky ReLU 0.1.  NJ = active 16-wide N tiles.
// ---------------------------------------------------------------------------
template <int NJ>
__global__ void k_conv3x3_wmma(
    const _Float16* __restrict__ act0, int Ca0,
    const _Float16* __restrict__ act1, int Ca1,
    int H, int W,
    const _Float16* __restrict__ wfrag, const float* __restrict__ bias,
    _Float16* __restrict__ outh, float* __restrict__ outf,
    int Cout, int Nt, int ntOff, int dil, int act)
{
    const int Cin = Ca0 + Ca1;         // multiple of 32
    const int Kc  = Cin >> 5;
    const int HW  = H * W;
    const int P   = BATCH * HW;

    const int L   = threadIdx.x;       // 0..31, wave32
    const int m   = L & 15;            // A row within M-tile
    const int hi  = L >> 4;            // K half selector
    const int mt  = blockIdx.x;
    const int nt0 = blockIdx.y * 4 + ntOff;

    const int pix = mt * 16 + m;
    const bool pvalid = pix < P;
    int b = pix / HW;
    int rem = pix - b * HW;
    int y = rem / W;
    int x = rem - y * W;

    v8f acc0 = {}, acc1 = {}, acc2 = {}, acc3 = {};

    for (int ky = 0; ky < 3; ++ky) {
        const int yy = y + (ky - 1) * dil;
        const bool yok = pvalid && (unsigned)yy < (unsigned)H;
        const long rowbase = ((long)b * H + yy) * W;
        for (int kx = 0; kx < 3; ++kx) {
            const int xx = x + (kx - 1) * dil;
            const bool inb = yok && (unsigned)xx < (unsigned)W;
            const long rowidx = rowbase + xx;
            const int tap = ky * 3 + kx;
            const _Float16* wtap = wfrag
                + ((size_t)tap * Kc) * (size_t)Nt * 512
                + (size_t)nt0 * 512 + (size_t)L * 16;

            for (int kc = 0; kc < Kc; ++kc) {
                const int c0 = kc << 5;    // wave-uniform chunk base channel
                v16h a = {};
                if (inb) {
                    const _Float16* ap = (c0 < Ca0)
                        ? act0 + rowidx * Ca0 + c0
                        : act1 + rowidx * Ca1 + (c0 - Ca0);
                    v8h alo = *(const v8h*)(ap + hi * 8);
                    v8h ahi = *(const v8h*)(ap + 16 + hi * 8);
                    a = __builtin_shufflevector(alo, ahi,
                            0, 1, 2, 3, 4, 5, 6, 7, 8, 9, 10, 11, 12, 13, 14, 15);
                }
                // Hoist all B-fragment loads ahead of the WMMA sequence so the
                // backend can clause them and drain with partial loadcnt waits.
                const _Float16* wk = wtap + (size_t)kc * Nt * 512;
                v16h bf0 = ldfrag(wk);
                v16h bf1 = {}, bf2 = {}, bf3 = {};
                if (NJ > 1) bf1 = ldfrag(wk + 512);
                if (NJ > 2) bf2 = ldfrag(wk + 1024);
                if (NJ > 3) bf3 = ldfrag(wk + 1536);

                acc0 = __builtin_amdgcn_wmma_f32_16x16x32_f16(
                    false, a, false, bf0, (short)0, acc0, false, false);
                if (NJ > 1)
                    acc1 = __builtin_amdgcn_wmma_f32_16x16x32_f16(
                        false, a, false, bf1, (short)0, acc1, false, false);
                if (NJ > 2)
                    acc2 = __builtin_amdgcn_wmma_f32_16x16x32_f16(
                        false, a, false, bf2, (short)0, acc2, false, false);
                if (NJ > 3)
                    acc3 = __builtin_amdgcn_wmma_f32_16x16x32_f16(
                        false, a, false, bf3, (short)0, acc3, false, false);
            }
        }
    }

    conv_store(acc0, nt0 + 0, L, mt, P, Cout, bias, outh, outf, act);
    if (NJ > 1) conv_store(acc1, nt0 + 1, L, mt, P, Cout, bias, outh, outf, act);
    if (NJ > 2) conv_store(acc2, nt0 + 2, L, mt, P, Cout, bias, outh, outf, act);
    if (NJ > 3) conv_store(acc3, nt0 + 3, L, mt, P, Cout, bias, outh, outf, act);
}

// Pack fp32 HWIO weights (3,3,Cin,Cout) into f16 WMMA B-fragment order:
// [tap][kc][nt][lane][e], zero-padded to Kc*32 x Nt*16.
__global__ void k_pack_wfrag(const float* __restrict__ w, _Float16* __restrict__ wp,
                             int Cin, int Cout, int Kc, int Nt)
{
    const size_t idx = (size_t)blockIdx.x * blockDim.x + threadIdx.x;
    const size_t tot = (size_t)9 * Kc * Nt * 512;
    if (idx >= tot) return;
    const int e    = (int)(idx & 15);
    const int lane = (int)((idx >> 4) & 31);
    const int t    = (int)(idx >> 9);       // tap*Kc*Nt + kc*Nt + nt
    const int nt   = t % Nt;
    const int t2   = t / Nt;
    const int kc   = t2 % Kc;
    const int tap  = t2 / Kc;
    const int hi = lane >> 4, n = lane & 15;
    const int k  = kc * 32 + wmma_kmap(e, hi);
    const int co = nt * 16 + n;
    float v = (k < Cin && co < Cout) ? w[((size_t)tap * Cin + k) * Cout + co] : 0.0f;
    wp[idx] = (_Float16)v;
}

// Virtual-concat 3x fp32 NHWC -> packed f16 [P][Cinp] (zero padded).
__global__ void k_pack_act3(const float* __restrict__ s0, int C0,
                            const float* __restrict__ s1, int C1,
                            const float* __restrict__ s2, int C2,
                            _Float16* __restrict__ o, int Cinp, int P)
{
    const size_t idx = (size_t)blockIdx.x * blockDim.x + threadIdx.x;
    if (idx >= (size_t)P * Cinp) return;
    const int pix = (int)(idx / Cinp);
    const int c   = (int)(idx % Cinp);
    float v = 0.0f;
    if (c < C0)                v = s0[(size_t)pix * C0 + c];
    else if (c < C0 + C1)      v = s1[(size_t)pix * C1 + (c - C0)];
    else if (c < C0 + C1 + C2) v = s2[(size_t)pix * C2 + (c - C0 - C1)];
    o[idx] = (_Float16)v;
}

// Context-net input: concat([flow2_raw f32 (1ch), o5 f16 (32ch)]) -> f16 [P][64].
__global__ void k_pack_act_cn(const float* __restrict__ flow,
                              const _Float16* __restrict__ o5,
                              _Float16* __restrict__ o, int P)
{
    const int idx = blockIdx.x * blockDim.x + threadIdx.x;
    if (idx >= P * 64) return;
    const int pix = idx >> 6;
    const int c   = idx & 63;
    float v = 0.0f;
    if (c == 0)      v = flow[pix];
    else if (c < 33) v = (float)o5[(size_t)pix * 32 + (c - 1)];
    o[idx] = (_Float16)v;
}

// Cost volume along x, 9 displacements, mean over C, leaky ReLU 0.1.
__global__ void k_costvol(const float* __restrict__ f1, const float* __restrict__ f2,
                          float* __restrict__ cv, int H, int W, int C)
{
    const int pix = blockIdx.x * blockDim.x + threadIdx.x;
    const int P = BATCH * H * W;
    if (pix >= P) return;
    const int x = pix % W;
    const float* a = f1 + (size_t)pix * C;
    const float invC = 1.0f / (float)C;
    for (int j = 0; j < 9; ++j) {
        const int xs = x + j - 4;
        float s = 0.0f;
        if ((unsigned)xs < (unsigned)W) {
            const float* bp = f2 + (size_t)(pix + (xs - x)) * C;
            for (int c = 0; c < C; ++c) s += a[c] * bp[c];
            s *= invC;
        }
        cv[(size_t)pix * 9 + j] = (s > 0.0f) ? s : 0.1f * s;
    }
}

// Inverse warp along x: out(x) = feat(x + flow*scale), bilinear, zero outside.
__global__ void k_warpx(const float* __restrict__ feat, const float* __restrict__ flow,
                        float scale, float* __restrict__ out, int H, int W, int C)
{
    const int pix = blockIdx.x * blockDim.x + threadIdx.x;
    const int P = BATCH * H * W;
    if (pix >= P) return;
    const int x = pix % W;
    const float xs = (float)x + flow[pix] * scale;
    const float x0 = floorf(xs);
    const float w1 = xs - x0;
    int x0i = (int)x0;
    x0i = x0i < 0 ? 0 : (x0i > W - 1 ? W - 1 : x0i);
    int x1i = x0i + 1;
    x1i = x1i > W - 1 ? W - 1 : x1i;
    const bool valid = (xs >= 0.0f) && (xs <= (float)(W - 1));
    const float* r0 = feat + (size_t)(pix - x + x0i) * C;
    const float* r1 = feat + (size_t)(pix - x + x1i) * C;
    float* o = out + (size_t)pix * C;
    for (int c = 0; c < C; ++c)
        o[c] = valid ? ((1.0f - w1) * r0[c] + w1 * r1[c]) : 0.0f;
}

// 2x bilinear upsample, single channel, half-pixel centers.
__global__ void k_up2(const float* __restrict__ src, float* __restrict__ dst,
                      int Hs, int Ws)
{
    const int Hd = Hs * 2, Wd = Ws * 2;
    const int pix = blockIdx.x * blockDim.x + threadIdx.x;
    const int P = BATCH * Hd * Wd;
    if (pix >= P) return;
    const int b = pix / (Hd * Wd);
    const int rem = pix % (Hd * Wd);
    const int y = rem / Wd, x = rem % Wd;
    const float sy = (y + 0.5f) * 0.5f - 0.5f;
    const float sx = (x + 0.5f) * 0.5f - 0.5f;
    int y0 = (int)floorf(sy), x0 = (int)floorf(sx);
    const float fy = sy - (float)y0, fx = sx - (float)x0;
    int y1 = y0 + 1, x1 = x0 + 1;
    y0 = y0 < 0 ? 0 : (y0 > Hs - 1 ? Hs - 1 : y0);
    y1 = y1 < 0 ? 0 : (y1 > Hs - 1 ? Hs - 1 : y1);
    x0 = x0 < 0 ? 0 : (x0 > Ws - 1 ? Ws - 1 : x0);
    x1 = x1 < 0 ? 0 : (x1 > Ws - 1 ? Ws - 1 : x1);
    const float* s = src + (size_t)b * Hs * Ws;
    const float v00 = s[y0 * Ws + x0], v01 = s[y0 * Ws + x1];
    const float v10 = s[y1 * Ws + x0], v11 = s[y1 * Ws + x1];
    dst[pix] = (1.0f - fy) * ((1.0f - fx) * v00 + fx * v01) +
               fy * ((1.0f - fx) * v10 + fx * v11);
}

__global__ void k_add_relu(const float* __restrict__ a, const float* __restrict__ b,
                           float* __restrict__ o, int n)
{
    const int i = blockIdx.x * blockDim.x + threadIdx.x;
    if (i >= n) return;
    float v = a[i] + (b ? b[i] : 0.0f);
    o[i] = v > 0.0f ? v : 0.0f;
}

__global__ void k_scale(const float* __restrict__ a, float s, float* __restrict__ o, int n)
{
    const int i = blockIdx.x * blockDim.x + threadIdx.x;
    if (i >= n) return;
    o[i] = a[i] * s;
}

// ---------------------------------------------------------------------------
// Host orchestration
// ---------------------------------------------------------------------------
extern "C" void kernel_launch(void* const* d_in, const int* in_sizes, int n_in,
                              void* d_out, int out_size, void* d_ws, size_t ws_size,
                              hipStream_t stream)
{
    (void)in_sizes; (void)n_in; (void)out_size; (void)ws_size;

    // ---- input index map -------------------------------------------------
    // d_in[0..4]  = f1_2..f1_6   d_in[5..9] = f2_2..f2_6
    // params pytree (sorted keys): cn(c1..c7: b,w) @10..23,
    // dec2 @24..35, dec3 @36..47, dec4 @48..59, dec5 @60..71, dec6 @72..83.
    const int cnBase = 10;
    const int decBase[5]  = {72, 60, 48, 36, 24};          // levels 6,5,4,3,2
    const int decCin1[5]  = {9, 138, 106, 74, 42};         // dec c1 real Cin
    const int decCinK[6]  = {0, 128, 256, 224, 160, 96};   // c2..c6 Cin
    const int decCout[6]  = {128, 128, 96, 64, 32, 1};
    const int cnCin[7]    = {33, 128, 128, 128, 96, 64, 32};
    const int cnCout[7]   = {128, 128, 128, 96, 64, 32, 1};
    const int cnDil[7]    = {1, 2, 4, 8, 16, 1, 1};

    const int Hs[5]  = {4, 8, 16, 32, 64};
    const int Wsz[5] = {13, 26, 52, 104, 208};
    const int Cs[5]  = {196, 128, 96, 64, 32};
    const float wsc[5] = {0.0f, 0.65f, 1.25f, 2.5f, 5.0f};
    int Pl[5], Cinp1[5];
    for (int i = 0; i < 5; ++i) {
        Pl[i] = BATCH * Hs[i] * Wsz[i];
        Cinp1[i] = (decCin1[i] + 31) & ~31;
    }

    const float* f1in[5] = {(const float*)d_in[4], (const float*)d_in[3],
                            (const float*)d_in[2], (const float*)d_in[1],
                            (const float*)d_in[0]};
    const float* f2in[5] = {(const float*)d_in[9], (const float*)d_in[8],
                            (const float*)d_in[7], (const float*)d_in[6],
                            (const float*)d_in[5]};

    // ---- workspace bump allocator ---------------------------------------
    char* ws = (char*)d_ws;
    size_t off = 0;
    auto alloc = [&](size_t bytes) -> char* {
        char* p = ws + off;
        off = (off + bytes + 255) & ~(size_t)255;
        return p;
    };

    // ---- pack all weights into fragment order ----------------------------
    auto packW = [&](const float* w, int Cin, int Cout) -> _Float16* {
        const int Kc = ((Cin + 31) & ~31) >> 5;
        const int Nt = (Cout + 15) / 16;
        const size_t tot = (size_t)9 * Kc * Nt * 512;
        _Float16* wp = (_Float16*)alloc(tot * sizeof(_Float16));
        const int th = 256;
        const int bl = (int)((tot + th - 1) / th);
        k_pack_wfrag<<<dim3(bl), dim3(th), 0, stream>>>(w, wp, Cin, Cout, Kc, Nt);
        return wp;
    };

    _Float16* wpDec[5][6];
    _Float16* wpCn[7];
    for (int li = 0; li < 5; ++li)
        for (int ci = 0; ci < 6; ++ci) {
            const int Cin = (ci == 0) ? decCin1[li] : decCinK[ci];
            wpDec[li][ci] = packW((const float*)d_in[decBase[li] + 2 * ci + 1],
                                  Cin, decCout[ci]);
        }
    for (int ci = 0; ci < 7; ++ci)
        wpCn[ci] = packW((const float*)d_in[cnBase + 2 * ci + 1], cnCin[ci], cnCout[ci]);

    // ---- persistent flow buffers (f32) -----------------------------------
    float* flowPers[5];   // [0]=flow6 .. [3]=flow3, [4]=flow2_raw
    for (int i = 0; i < 5; ++i) flowPers[i] = (float*)alloc((size_t)Pl[i] * 4);
    float* flow2 = (float*)alloc((size_t)Pl[4] * 4);
    float* cnout = (float*)alloc((size_t)Pl[4] * 4);

    // ---- shared level arena (reused across levels) -----------------------
    size_t arenaBytes = 0;
    for (int i = 0; i < 5; ++i) {
        size_t f32b = (size_t)Pl[i] * (1 + Cs[i] + 9 + 1) * 4;   // flowup,f2w,cv,fdec
        size_t f16b = (size_t)Pl[i] * (Cinp1[i] + 128 + 128 + 96 + 64 + 32 + 64) * 2;
        size_t s = f32b + f16b + 16 * 256;
        if (s > arenaBytes) arenaBytes = s;
    }
    char* arena = alloc(arenaBytes);

    // Launch one NJ-specialized sub-launch of the conv kernel.
    auto convPart = [&](int NJ, int gridY, int ntOff,
                        const _Float16* a0, int C0, const _Float16* a1, int C1,
                        int H, int W, _Float16* wf, const float* bias,
                        int Cout, int Nt, int dil, int act,
                        _Float16* oh, float* of) {
        const int P = BATCH * H * W;
        dim3 grid((P + 15) / 16, gridY);
        switch (NJ) {
        case 4: k_conv3x3_wmma<4><<<grid, dim3(32), 0, stream>>>(
                    a0, C0, a1, C1, H, W, wf, bias, oh, of, Cout, Nt, ntOff, dil, act);
                break;
        case 3: k_conv3x3_wmma<3><<<grid, dim3(32), 0, stream>>>(
                    a0, C0, a1, C1, H, W, wf, bias, oh, of, Cout, Nt, ntOff, dil, act);
                break;
        case 2: k_conv3x3_wmma<2><<<grid, dim3(32), 0, stream>>>(
                    a0, C0, a1, C1, H, W, wf, bias, oh, of, Cout, Nt, ntOff, dil, act);
                break;
        default: k_conv3x3_wmma<1><<<grid, dim3(32), 0, stream>>>(
                    a0, C0, a1, C1, H, W, wf, bias, oh, of, Cout, Nt, ntOff, dil, act);
                break;
        }
    };

    auto conv = [&](const _Float16* a0, int C0, const _Float16* a1, int C1,
                    int H, int W, _Float16* wf, const float* bias,
                    int Cout, int dil, int act, _Float16* oh, float* of) {
        const int Nt = (Cout + 15) / 16;
        const int full = Nt / 4, rem = Nt % 4;
        if (full > 0)
            convPart(4, full, 0, a0, C0, a1, C1, H, W, wf, bias,
                     Cout, Nt, dil, act, oh, of);
        if (rem > 0)
            convPart(rem, 1, full * 4, a0, C0, a1, C1, H, W, wf, bias,
                     Cout, Nt, dil, act, oh, of);
    };

    const int th = 256;

    for (int li = 0; li < 5; ++li) {
        const int H = Hs[li], W = Wsz[li], C = Cs[li], P = Pl[li];
        const int Cp1 = Cinp1[li];
        size_t aoff = 0;
        auto la = [&](size_t bytes) -> char* {
            char* p = arena + aoff;
            aoff = (aoff + bytes + 255) & ~(size_t)255;
            return p;
        };
        float*     flowup = (float*)la((size_t)P * 4);
        float*     f2w    = (float*)la((size_t)P * C * 4);
        float*     cv     = (float*)la((size_t)P * 9 * 4);
        float*     fdec   = (float*)la((size_t)P * 4);
        _Float16*  x1     = (_Float16*)la((size_t)P * Cp1 * 2);
        _Float16*  o1     = (_Float16*)la((size_t)P * 128 * 2);
        _Float16*  o2     = (_Float16*)la((size_t)P * 128 * 2);
        _Float16*  o3     = (_Float16*)la((size_t)P * 96 * 2);
        _Float16*  o4     = (_Float16*)la((size_t)P * 64 * 2);
        _Float16*  o5     = (_Float16*)la((size_t)P * 32 * 2);
        _Float16*  cnx    = (_Float16*)la((size_t)P * 64 * 2);

        const float* f1 = f1in[li];
        const float* f2 = f2in[li];
        const int blocksP = (P + th - 1) / th;

        if (li == 0) {
            k_costvol<<<blocksP, th, 0, stream>>>(f1, f2, cv, H, W, C);
            const size_t nx = (size_t)P * Cp1;
            k_pack_act3<<<(int)((nx + th - 1) / th), th, 0, stream>>>(
                cv, 9, nullptr, 0, nullptr, 0, x1, Cp1, P);
        } else {
            k_up2<<<blocksP, th, 0, stream>>>(flowPers[li - 1], flowup, H / 2, W / 2);
            k_warpx<<<blocksP, th, 0, stream>>>(f2, flowup, wsc[li], f2w, H, W, C);
            k_costvol<<<blocksP, th, 0, stream>>>(f1, f2w, cv, H, W, C);
            const size_t nx = (size_t)P * Cp1;
            k_pack_act3<<<(int)((nx + th - 1) / th), th, 0, stream>>>(
                cv, 9, f1, C, flowup, 1, x1, Cp1, P);
        }

        const int base = decBase[li];
        auto DB = [&](int ci) -> const float* {
            return (const float*)d_in[base + 2 * ci];
        };

        conv(x1, Cp1, nullptr, 0, H, W, wpDec[li][0], DB(0), 128, 1, 1, o1, nullptr);
        conv(o1, 128, nullptr, 0, H, W, wpDec[li][1], DB(1), 128, 1, 1, o2, nullptr);
        conv(o1, 128, o2, 128, H, W, wpDec[li][2], DB(2), 96, 1, 1, o3, nullptr);
        conv(o2, 128, o3, 96,  H, W, wpDec[li][3], DB(3), 64, 1, 1, o4, nullptr);
        conv(o3, 96,  o4, 64,  H, W, wpDec[li][4], DB(4), 32, 1, 1, o5, nullptr);
        conv(o4, 64,  o5, 32,  H, W, wpDec[li][5], DB(5), 1, 1, 0, nullptr, fdec);

        k_add_relu<<<blocksP, th, 0, stream>>>(
            fdec, (li == 0) ? nullptr : flowup, flowPers[li], P);

        if (li == 4) {
            // Context network on concat([flow2_raw, o5]); ping-pong via o1/o2.
            auto CB = [&](int ci) -> const float* {
                return (const float*)d_in[cnBase + 2 * ci];
            };
            k_pack_act_cn<<<(P * 64 + th - 1) / th, th, 0, stream>>>(
                flowPers[4], o5, cnx, P);
            conv(cnx, 64, nullptr, 0, H, W, wpCn[0], CB(0), 128, cnDil[0], 1, o1, nullptr);
            conv(o1, 128, nullptr, 0, H, W, wpCn[1], CB(1), 128, cnDil[1], 1, o2, nullptr);
            conv(o2, 128, nullptr, 0, H, W, wpCn[2], CB(2), 128, cnDil[2], 1, o1, nullptr);
            conv(o1, 128, nullptr, 0, H, W, wpCn[3], CB(3), 96, cnDil[3], 1, o2, nullptr);
            conv(o2, 96,  nullptr, 0, H, W, wpCn[4], CB(4), 64, cnDil[4], 1, o1, nullptr);
            conv(o1, 64,  nullptr, 0, H, W, wpCn[5], CB(5), 32, cnDil[5], 1, o2, nullptr);
            conv(o2, 32,  nullptr, 0, H, W, wpCn[6], CB(6), 1, cnDil[6], 0, nullptr, cnout);
            k_add_relu<<<blocksP, th, 0, stream>>>(cnout, flowPers[4], flow2, P);
        }
    }

    // ---- write scaled disparities to d_out -------------------------------
    float* out = (float*)d_out;
    const int P2 = Pl[4], P3 = Pl[3], P4 = Pl[2], P5 = Pl[1];
    k_scale<<<(P2 + th - 1) / th, th, 0, stream>>>(flow2, 5.0f / 208.0f, out, P2);
    k_scale<<<(P3 + th - 1) / th, th, 0, stream>>>(flowPers[3], 2.5f / 104.0f, out + P2, P3);
    k_scale<<<(P4 + th - 1) / th, th, 0, stream>>>(flowPers[2], 1.25f / 52.0f, out + P2 + P3, P4);
    k_scale<<<(P5 + th - 1) / th, th, 0, stream>>>(flowPers[1], 0.65f / 26.0f, out + P2 + P3 + P4, P5);
}
